// Attention_738734375125
// MI455X (gfx1250) — compile-verified, matrix operand order checked by reading, not run on hardware
//
#include <hip/hip_runtime.h>

// ---------------------------------------------------------------------------
// RWKV-ish attention for MI455X (gfx1250, wave32, WMMA bf16, async->LDS).
// B=4, T=4096, D=1024.  M = B*T = 16384 GEMM rows.
// ---------------------------------------------------------------------------

#define B_DIM 4
#define T_DIM 4096
#define D_DIM 1024
#define M_DIM (B_DIM * T_DIM)   // 16384

typedef __bf16 bf16_t;
typedef __attribute__((ext_vector_type(16))) __bf16 v16bf;
typedef __attribute__((ext_vector_type(8)))  __bf16 v8bf;
typedef __attribute__((ext_vector_type(4)))  __bf16 v4bf;
typedef __attribute__((ext_vector_type(8)))  float  v8f;

#define LDS_K 40   // 32 bf16 + 8 pad -> 80B row stride, conflict-free b128 frag loads

// --- CDNA5 async global->LDS copy (ASYNCcnt-tracked, no VGPR data path) ----
__device__ __forceinline__ void async_b128(unsigned lds_addr, const bf16_t* gbase,
                                           unsigned goff_bytes) {
    asm volatile("global_load_async_to_lds_b128 %0, %1, %2 offset:0"
                 :: "v"(lds_addr), "v"(goff_bytes), "s"(gbase)
                 : "memory");
}
__device__ __forceinline__ void wait_async0() {
    asm volatile("s_wait_asynccnt 0" ::: "memory");
}
__device__ __forceinline__ unsigned lds_off(const void* p) {
    return (unsigned)(uintptr_t)p;   // generic LDS addr low 32 bits == LDS offset
}

// Load a 16x32 bf16 WMMA operand fragment from a row-major LDS tile.
// ISA 7.12.2 (16-bit A 16x32): lane (idx, half=0) holds row idx, K {0..7}U{16..23};
// half=1 lanes hold K {8..15}U{24..31}.  B (32x16) is the symmetric layout.
__device__ __forceinline__ v16bf load_frag(const bf16_t* tile, int row0, int idx, int half) {
    const bf16_t* p = tile + (row0 + idx) * LDS_K + half * 8;
    v8bf lo = *(const v8bf*)(p);
    v8bf hi = *(const v8bf*)(p + 16);
    return __builtin_shufflevector(lo, hi, 0,1,2,3,4,5,6,7,8,9,10,11,12,13,14,15);
}

// ---------------------------------------------------------------------------
// Cast x (fp32) -> xbf (bf16), one float4 per thread.
// ---------------------------------------------------------------------------
__global__ __launch_bounds__(256) void cast_kernel(
    const float* __restrict__ x, bf16_t* __restrict__ xbf)
{
    const size_t i = ((size_t)blockIdx.x * 256 + threadIdx.x) * 4;
    const float4 f = *(const float4*)(x + i);
    v4bf p = { (bf16_t)f.x, (bf16_t)f.y, (bf16_t)f.z, (bf16_t)f.w };
    *(v4bf*)(xbf + i) = p;
}

// ---------------------------------------------------------------------------
// Prep: fold time_mix into weights (bf16), compute per-output biases
//   Wp[e,d] = tm[d]*W[e,d]    c[e] = sum_d (1-tm[d])*lx[d]*W[e,d]
// ---------------------------------------------------------------------------
__global__ __launch_bounds__(256) void prep_kernel(
    const float* __restrict__ Wk, const float* __restrict__ Wv,
    const float* __restrict__ Wr, const float* __restrict__ Wo,
    const float* __restrict__ tmk, const float* __restrict__ tmv,
    const float* __restrict__ tmr, const float* __restrict__ lx,
    bf16_t* __restrict__ Wkp, bf16_t* __restrict__ Wvp,
    bf16_t* __restrict__ Wrp, bf16_t* __restrict__ Wop,
    float* __restrict__ ck, float* __restrict__ cv, float* __restrict__ cr)
{
    const int e   = blockIdx.x;
    const int wid = blockIdx.y;
    const int tid = threadIdx.x;

    const float* W  = (wid == 0) ? Wk  : (wid == 1) ? Wv  : (wid == 2) ? Wr  : Wo;
    const float* tm = (wid == 0) ? tmk : (wid == 1) ? tmv : (wid == 2) ? tmr : nullptr;
    bf16_t*      Wp = (wid == 0) ? Wkp : (wid == 1) ? Wvp : (wid == 2) ? Wrp : Wop;
    float*       c  = (wid == 0) ? ck  : (wid == 1) ? cv  : (wid == 2) ? cr  : nullptr;

    float partial = 0.0f;
    for (int d = tid; d < D_DIM; d += 256) {
        float w = W[(size_t)e * D_DIM + d];
        float t = tm ? tm[d] : 1.0f;
        Wp[(size_t)e * D_DIM + d] = (bf16_t)(w * t);
        if (c) partial += (1.0f - t) * lx[d] * w;
    }
    __shared__ float red[256];
    red[tid] = partial;
    __syncthreads();
    for (int s = 128; s > 0; s >>= 1) {
        if (tid < s) red[tid] += red[tid + s];
        __syncthreads();
    }
    if (tid == 0 && c) c[e] = red[0];
}

// ---------------------------------------------------------------------------
// Fused k/v/r GEMM + WKV epilogue, double-buffered async->LDS staging.
// Block tile 128(M) x 64(N); 8 waves (4x2), wave tile 32x32;
// 12 WMMAs / wave / k-step; 5 async_b128 / thread / k-step.
// ---------------------------------------------------------------------------
__global__ __launch_bounds__(256) void kvr_wkv_kernel(
    const bf16_t* __restrict__ xbf,
    const bf16_t* __restrict__ Wkp, const bf16_t* __restrict__ Wvp,
    const bf16_t* __restrict__ Wrp,
    const float* __restrict__ ck, const float* __restrict__ cv,
    const float* __restrict__ cr, const float* __restrict__ tf,
    const float* __restrict__ lnum, const float* __restrict__ lden,
    bf16_t* __restrict__ rw)
{
    __shared__ alignas(16) bf16_t As[2][128 * LDS_K];
    __shared__ alignas(16) bf16_t Ws[2][3][64 * LDS_K];

    const int tid  = threadIdx.x;
    const int lane = tid & 31;
    const int wave = tid >> 5;
    const int idx  = lane & 15;
    const int half = lane >> 4;
    const int wm   = wave >> 1;   // 0..3 -> rows
    const int wn   = wave & 1;    // 0..1 -> cols

    const int n0 = blockIdx.x * 64;
    const int m0 = blockIdx.y * 128;

    const bf16_t* xblk = xbf + (size_t)m0 * D_DIM;
    const bf16_t* wblk[3] = { Wkp + (size_t)n0 * D_DIM,
                              Wvp + (size_t)n0 * D_DIM,
                              Wrp + (size_t)n0 * D_DIM };

    // Per-thread staging coordinates (16B chunks): A: 512 chunks (2/thread),
    // each W tile: 256 chunks (1/thread).
    const int arow0 = tid >> 2,          ac8 = tid & 3;          // rows 0..63
    const int arow1 = (tid + 256) >> 2;                          // rows 64..127
    const unsigned a_g0 = (unsigned)((arow0 * D_DIM + ac8 * 8) * 2);
    const unsigned a_g1 = (unsigned)((arow1 * D_DIM + ac8 * 8) * 2);
    const unsigned a_l0 = (unsigned)((arow0 * LDS_K + ac8 * 8) * 2);
    const unsigned a_l1 = (unsigned)((arow1 * LDS_K + ac8 * 8) * 2);
    const int wrow = tid >> 2,           wc8 = tid & 3;          // rows 0..63
    const unsigned w_g = (unsigned)((wrow * D_DIM + wc8 * 8) * 2);
    const unsigned w_l = (unsigned)((wrow * LDS_K + wc8 * 8) * 2);

    const unsigned asb[2]  = { lds_off(&As[0][0]), lds_off(&As[1][0]) };
    const unsigned wsb[2][3] = {
        { lds_off(&Ws[0][0][0]), lds_off(&Ws[0][1][0]), lds_off(&Ws[0][2][0]) },
        { lds_off(&Ws[1][0][0]), lds_off(&Ws[1][1][0]), lds_off(&Ws[1][2][0]) } };

    auto issue = [&](int buf, int k0) {
        const unsigned kb = (unsigned)(k0 * 2);
        async_b128(asb[buf] + a_l0, xblk, a_g0 + kb);
        async_b128(asb[buf] + a_l1, xblk, a_g1 + kb);
#pragma unroll
        for (int w = 0; w < 3; w++)
            async_b128(wsb[buf][w] + w_l, wblk[w], w_g + kb);
    };

    v8f acc[3][4];
#pragma unroll
    for (int w = 0; w < 3; w++)
#pragma unroll
        for (int f = 0; f < 4; f++) acc[w][f] = (v8f)(0.0f);

    issue(0, 0);

    const int NK = D_DIM / 32;   // 32 k-steps
    for (int it = 0; it < NK; ++it) {
        const int cur = it & 1;
        wait_async0();           // own async loads for tile `it` complete
        __syncthreads();         // all waves' tile data visible; prev compute done
        if (it + 1 < NK) issue(cur ^ 1, (it + 1) * 32);   // overlap with compute

        const bf16_t* Atile = As[cur];
        v16bf a0 = load_frag(Atile, wm * 32,      idx, half);
        v16bf a1 = load_frag(Atile, wm * 32 + 16, idx, half);
#pragma unroll
        for (int w = 0; w < 3; w++) {
            const bf16_t* Wtile = Ws[cur][w];
            v16bf b0 = load_frag(Wtile, wn * 32,      idx, half);
            v16bf b1 = load_frag(Wtile, wn * 32 + 16, idx, half);
            acc[w][0] = __builtin_amdgcn_wmma_f32_16x16x32_bf16(false, a0, false, b0, (short)0, acc[w][0], false, false);
            acc[w][1] = __builtin_amdgcn_wmma_f32_16x16x32_bf16(false, a0, false, b1, (short)0, acc[w][1], false, false);
            acc[w][2] = __builtin_amdgcn_wmma_f32_16x16x32_bf16(false, a1, false, b0, (short)0, acc[w][2], false, false);
            acc[w][3] = __builtin_amdgcn_wmma_f32_16x16x32_bf16(false, a1, false, b1, (short)0, acc[w][3], false, false);
        }
    }

    // --- WKV epilogue: D-matrix layout: VGPR r -> M = r + half*8, N = idx ----
#pragma unroll
    for (int nf = 0; nf < 2; nf++) {
        const int e   = n0 + wn * 32 + nf * 16 + idx;
        const float ckv = ck[e], cvv = cv[e], crv = cr[e];
        const float tfv = tf[e], ln = lnum[e], ld = lden[e];
#pragma unroll
        for (int mf = 0; mf < 2; mf++) {
            const int f     = mf * 2 + nf;
            const int mbase = m0 + wm * 32 + mf * 16 + half * 8;
#pragma unroll
            for (int r = 0; r < 8; r++) {
                float kk = acc[0][f][r] + ckv;
                float vv = acc[1][f][r] + cvv;
                float rr = acc[2][f][r] + crv;
                float efk  = __expf(tfv + kk);
                float wkv  = (ln + efk * vv) / (ld + efk);
                float sig  = 1.0f / (1.0f + __expf(-rr));
                rw[(size_t)(mbase + r) * D_DIM + e] = (bf16_t)(sig * wkv);
            }
        }
    }
}

// ---------------------------------------------------------------------------
// Output GEMM: out(fp32) = rwkv(bf16) @ Wo^T(bf16), async double-buffered.
// Block 128x128, 8 waves (4x2), wave tile 32x64 => 8 WMMAs / wave / k-step.
// ---------------------------------------------------------------------------
__global__ __launch_bounds__(256) void out_gemm_kernel(
    const bf16_t* __restrict__ A, const bf16_t* __restrict__ W,
    float* __restrict__ out)
{
    __shared__ alignas(16) bf16_t As[2][128 * LDS_K];
    __shared__ alignas(16) bf16_t Bs[2][128 * LDS_K];

    const int tid  = threadIdx.x;
    const int lane = tid & 31;
    const int wave = tid >> 5;
    const int idx  = lane & 15;
    const int half = lane >> 4;
    const int wm   = wave >> 1;   // 0..3
    const int wn   = wave & 1;    // 0..1

    const int n0 = blockIdx.x * 128;
    const int m0 = blockIdx.y * 128;

    const bf16_t* ablk = A + (size_t)m0 * D_DIM;
    const bf16_t* bblk = W + (size_t)n0 * D_DIM;

    const int row0 = tid >> 2,          c8 = tid & 3;      // rows 0..63
    const int row1 = (tid + 256) >> 2;                     // rows 64..127
    const unsigned g0 = (unsigned)((row0 * D_DIM + c8 * 8) * 2);
    const unsigned g1 = (unsigned)((row1 * D_DIM + c8 * 8) * 2);
    const unsigned l0 = (unsigned)((row0 * LDS_K + c8 * 8) * 2);
    const unsigned l1 = (unsigned)((row1 * LDS_K + c8 * 8) * 2);

    const unsigned asb[2] = { lds_off(&As[0][0]), lds_off(&As[1][0]) };
    const unsigned bsb[2] = { lds_off(&Bs[0][0]), lds_off(&Bs[1][0]) };

    auto issue = [&](int buf, int k0) {
        const unsigned kb = (unsigned)(k0 * 2);
        async_b128(asb[buf] + l0, ablk, g0 + kb);
        async_b128(asb[buf] + l1, ablk, g1 + kb);
        async_b128(bsb[buf] + l0, bblk, g0 + kb);
        async_b128(bsb[buf] + l1, bblk, g1 + kb);
    };

    v8f acc[8];
#pragma unroll
    for (int f = 0; f < 8; f++) acc[f] = (v8f)(0.0f);

    issue(0, 0);

    const int NK = D_DIM / 32;
    for (int it = 0; it < NK; ++it) {
        const int cur = it & 1;
        wait_async0();
        __syncthreads();
        if (it + 1 < NK) issue(cur ^ 1, (it + 1) * 32);

        v16bf a0 = load_frag(As[cur], wm * 32,      idx, half);
        v16bf a1 = load_frag(As[cur], wm * 32 + 16, idx, half);
#pragma unroll
        for (int nf = 0; nf < 4; nf++) {
            v16bf b = load_frag(Bs[cur], wn * 64 + nf * 16, idx, half);
            acc[nf]     = __builtin_amdgcn_wmma_f32_16x16x32_bf16(false, a0, false, b, (short)0, acc[nf],     false, false);
            acc[4 + nf] = __builtin_amdgcn_wmma_f32_16x16x32_bf16(false, a1, false, b, (short)0, acc[4 + nf], false, false);
        }
    }

#pragma unroll
    for (int mf = 0; mf < 2; mf++) {
        const int mbase = m0 + wm * 32 + mf * 16 + half * 8;
#pragma unroll
        for (int nf = 0; nf < 4; nf++) {
            const int e = n0 + wn * 64 + nf * 16 + idx;
#pragma unroll
            for (int r = 0; r < 8; r++)
                out[(size_t)(mbase + r) * D_DIM + e] = acc[mf * 4 + nf][r];
        }
    }
}

// ---------------------------------------------------------------------------
// State outputs for t = T-1 (tiny: B*D = 4096 elements, fp32 throughout).
// ---------------------------------------------------------------------------
__global__ __launch_bounds__(256) void state_kernel(
    const float* __restrict__ x,  const float* __restrict__ lx,
    const float* __restrict__ lnum, const float* __restrict__ lden,
    const float* __restrict__ tdecay,
    const float* __restrict__ tmk, const float* __restrict__ tmv,
    const float* __restrict__ Wk, const float* __restrict__ Wv,
    float* __restrict__ x_out, float* __restrict__ num_out, float* __restrict__ den_out)
{
    const int t = blockIdx.x * blockDim.x + threadIdx.x;
    if (t >= B_DIM * D_DIM) return;
    const int b = t >> 10;           // D = 1024
    const int e = t & (D_DIM - 1);

    const float* xrow = x + ((size_t)b * T_DIM + (T_DIM - 1)) * D_DIM;
    float kk = 0.0f, vv = 0.0f;
    for (int d = 0; d < D_DIM; d++) {
        float xd = xrow[d], lxd = lx[d];
        kk += (xd * tmk[d] + lxd * (1.0f - tmk[d])) * Wk[(size_t)e * D_DIM + d];
        vv += (xd * tmv[d] + lxd * (1.0f - tmv[d])) * Wv[(size_t)e * D_DIM + d];
    }
    const float ek    = __expf(kk);
    const float decay = __expf(-__expf(tdecay[e]));
    x_out[t]   = xrow[e];
    num_out[t] = decay * lnum[e] + ek * vv;
    den_out[t] = decay * lden[e] + ek;
}

// ---------------------------------------------------------------------------
extern "C" void kernel_launch(void* const* d_in, const int* in_sizes, int n_in,
                              void* d_out, int out_size, void* d_ws, size_t ws_size,
                              hipStream_t stream)
{
    const float* x      = (const float*)d_in[0];
    const float* lx     = (const float*)d_in[1];
    const float* lnum   = (const float*)d_in[2];
    const float* lden   = (const float*)d_in[3];
    const float* tdecay = (const float*)d_in[4];
    const float* tf     = (const float*)d_in[5];
    const float* tmk    = (const float*)d_in[6];
    const float* tmv    = (const float*)d_in[7];
    const float* tmr    = (const float*)d_in[8];
    const float* Wk     = (const float*)d_in[9];
    const float* Wv     = (const float*)d_in[10];
    const float* Wr     = (const float*)d_in[11];
    const float* Wo     = (const float*)d_in[12];

    float* out    = (float*)d_out;
    float* x_last = out + (size_t)M_DIM * D_DIM;
    float* num_o  = x_last + (size_t)B_DIM * D_DIM;
    float* den_o  = num_o + (size_t)B_DIM * D_DIM;

    // Workspace layout
    char* wsb = (char*)d_ws;
    const size_t WBYTES  = (size_t)D_DIM * D_DIM * sizeof(bf16_t);   // 2 MB
    const size_t RWBYTES = (size_t)M_DIM * D_DIM * sizeof(bf16_t);   // 32 MB
    bf16_t* Wkp = (bf16_t*)(wsb);
    bf16_t* Wvp = (bf16_t*)(wsb + WBYTES);
    bf16_t* Wrp = (bf16_t*)(wsb + 2 * WBYTES);
    bf16_t* Wop = (bf16_t*)(wsb + 3 * WBYTES);
    float*  ck  = (float*)(wsb + 4 * WBYTES);
    float*  cv  = ck + D_DIM;
    float*  cr  = cv + D_DIM;
    bf16_t* rw  = (bf16_t*)(wsb + 4 * WBYTES + 16384);
    bf16_t* xbf = (bf16_t*)(wsb + 4 * WBYTES + 16384 + RWBYTES);

    // 1) fold time_mix into bf16 weights + biases; cast x to bf16
    prep_kernel<<<dim3(D_DIM, 4), 256, 0, stream>>>(
        Wk, Wv, Wr, Wo, tmk, tmv, tmr, lx, Wkp, Wvp, Wrp, Wop, ck, cv, cr);
    cast_kernel<<<dim3((M_DIM * D_DIM) / (256 * 4)), 256, 0, stream>>>(x, xbf);

    // 2) fused k/v/r GEMMs + WKV epilogue -> rwkv (bf16)
    kvr_wkv_kernel<<<dim3(D_DIM / 64, M_DIM / 128), 256, 0, stream>>>(
        xbf, Wkp, Wvp, Wrp, ck, cv, cr, tf, lnum, lden, rw);

    // 3) out = rwkv @ Wo^T
    out_gemm_kernel<<<dim3(D_DIM / 128, M_DIM / 128), 256, 0, stream>>>(rw, Wop, out);

    // 4) (B,D) state outputs
    state_kernel<<<dim3((B_DIM * D_DIM + 255) / 256), 256, 0, stream>>>(
        x, lx, lnum, lden, tdecay, tmk, tmv, Wk, Wv, x_last, num_o, den_o);
}
